// SinkBF_multilayers_3401614098545
// MI455X (gfx1250) — matrix operand order repaired and checked by reading, not compiled, not run on hardware
//
#include <hip/hip_runtime.h>

// SinkBF multilayer GRU rollout for gfx1250 (MI455X).
// One workgroup (8 waves, 256 thr) owns a 16-row batch tile for all 512 steps.
// Weights live as per-wave WMMA B-fragments in VGPRs (bf16); hidden state in
// f32 accumulator-layout registers; LDS holds bf16 activation staging only.
// Nonlinearities use the CDNA5 hardware V_TANH_F32 (branch-free critical path).

typedef __attribute__((ext_vector_type(16))) __bf16 v16bf;
typedef __attribute__((ext_vector_type(8)))  float  v8f;

#define NXD   128
#define NWD   64
#define NYD   32
#define NB    1024
#define NWUP_ 256
#define TT    512          // NWUP + NHRZN

#define LDX 136            // padded bf16 row strides (16B-aligned, bank-rotating)
#define LDW 72
#define LDE 40

__device__ __forceinline__ v8f wmma_bf(v16bf a, v16bf b, v8f c) {
  return __builtin_amdgcn_wmma_f32_16x16x32_bf16(false, a, false, b, (short)0, c,
                                                 false, false);
}

// Branch-free tanh: hardware V_TANH_F32 if the builtin exists, else exp/rcp.
__device__ __forceinline__ float fast_tanh(float x) {
#if defined(__has_builtin)
#if __has_builtin(__builtin_amdgcn_tanhf)
  return __builtin_amdgcn_tanhf(x);
#else
  const float e = __expf(2.f * x);
  return 1.f - 2.f * __builtin_amdgcn_rcpf(e + 1.f);
#endif
#else
  const float e = __expf(2.f * x);
  return 1.f - 2.f * __builtin_amdgcn_rcpf(e + 1.f);
#endif
}
// sigmoid(x) = 0.5 * tanh(x/2) + 0.5  (single TRANS op, branch-free)
__device__ __forceinline__ float fast_sigmoid(float x) {
  return __builtin_fmaf(0.5f, fast_tanh(0.5f * x), 0.5f);
}

// A-fragment (16x32 bf16, row-major M x K in LDS). ISA 7.12.2 16-bit A layout:
// lane m=l&15, hi=l>>4; elems 0..7 -> K = k0+hi*8+j ; elems 8..15 -> K = k0+16+hi*8+j
__device__ __forceinline__ v16bf lds_afrag(const __bf16* buf, int ldk, int k0, int lane) {
  const int m  = lane & 15;
  const int hi = lane >> 4;
  const __bf16* p0 = buf + m * ldk + k0 + hi * 8;
  v16bf a;
#pragma unroll
  for (int j = 0; j < 8; ++j) { a[j] = p0[j]; a[8 + j] = p0[16 + j]; }
  return a;
}

// B-fragment (32x16 = K x N) for x @ W^T : B[k][n] = W[n][k], W row-major (N,K) f32.
// ISA B layout: lane n=l&15, hi=l>>4; elems 0..7 -> K = k0+hi*8+j ; elems 8..15 -> K = k0+16+hi*8+j
__device__ __forceinline__ v16bf gbl_bfrag(const float* W, int K, int n0, int k0, int lane) {
  const int n  = n0 + (lane & 15);
  const int kb = k0 + ((lane >> 4) << 3);
  const float* p = W + (size_t)n * K + kb;
  v16bf b;
#pragma unroll
  for (int j = 0; j < 8; ++j) { b[j] = (__bf16)p[j]; b[8 + j] = (__bf16)p[16 + j]; }
  return b;
}

__global__ __launch_bounds__(256, 1) void sinkbf_gru_rollout(
    const float* __restrict__ Y0,
    const float* __restrict__ Wq,   const float* __restrict__ bq,
    const float* __restrict__ Wp1,  const float* __restrict__ bp1,
    const float* __restrict__ Wp2,  const float* __restrict__ bp2,
    const float* __restrict__ Ww,   const float* __restrict__ bw,
    const float* __restrict__ Wih0, const float* __restrict__ Whh0,
    const float* __restrict__ bih0, const float* __restrict__ bhh0,
    const float* __restrict__ Wih1, const float* __restrict__ Whh1,
    const float* __restrict__ bih1, const float* __restrict__ bhh1,
    float* __restrict__ Xout, float* __restrict__ Wout, float* __restrict__ Yout) {
  __shared__ __align__(16) __bf16 sX0[16 * LDX];   // h0 state (bf16 copy)
  __shared__ __align__(16) __bf16 sX1[16 * LDX];   // h1 state (bf16 copy)
  __shared__ __align__(16) __bf16 sT [16 * LDX];   // tanh(h1 Wp1^T) staging
  __shared__ __align__(16) __bf16 sW [16 * LDW];   // w staging
  __shared__ __align__(16) __bf16 sE [16 * LDE];   // e (or Y0[0]) staging
  __shared__ float sYh[16 * NYD];                  // yhat f32 (for next e)

  const int tid  = threadIdx.x;
  const int lane = tid & 31;
  const int wv   = tid >> 5;     // 8 waves
  const int lnn  = lane & 15;
  const int hi   = lane >> 4;
  const int b0   = blockIdx.x * 16;
  const int c0   = wv * 16 + lnn;  // this wave's owned column block

  // ---------- register-resident weight fragments (f32 -> bf16 once) ----------
  v16bf fWih0[3][2], fWhh0[3][4], fWih1[3][4], fWhh1[3][4], fWp1[4];
#pragma unroll
  for (int i = 0; i < 3; ++i) {            // i=0: r-tile, 1: z-tile, 2: n-tile
    const int n0 = (wv + 8 * i) * 16;
#pragma unroll
    for (int kc = 0; kc < 2; ++kc) fWih0[i][kc] = gbl_bfrag(Wih0, NWD, n0, kc * 32, lane);
#pragma unroll
    for (int kc = 0; kc < 4; ++kc) {
      fWhh0[i][kc] = gbl_bfrag(Whh0, NXD, n0, kc * 32, lane);
      fWih1[i][kc] = gbl_bfrag(Wih1, NXD, n0, kc * 32, lane);
      fWhh1[i][kc] = gbl_bfrag(Whh1, NXD, n0, kc * 32, lane);
    }
  }
#pragma unroll
  for (int kc = 0; kc < 4; ++kc) fWp1[kc] = gbl_bfrag(Wp1, NXD, wv * 16, kc * 32, lane);

  v16bf fWw[5];                       // waves 0..3 own the 4 N-tiles of w
  if (wv < 4) {
#pragma unroll
    for (int kc = 0; kc < 5; ++kc) fWw[kc] = gbl_bfrag(Ww, NXD + NYD, wv * 16, kc * 32, lane);
  }
  v16bf fWp2[4];                      // waves 0..1 own the 2 N-tiles of yhat
  if (wv < 2) {
#pragma unroll
    for (int kc = 0; kc < 4; ++kc) fWp2[kc] = gbl_bfrag(Wp2, NXD, wv * 16, kc * 32, lane);
  }

  // ---------- per-lane biases (one column each) ----------
  const float bR0  = bih0[c0] + bhh0[c0];
  const float bZ0  = bih0[NXD + c0] + bhh0[NXD + c0];
  const float bIN0 = bih0[2 * NXD + c0];
  const float bHN0 = bhh0[2 * NXD + c0];
  const float bR1  = bih1[c0] + bhh1[c0];
  const float bZ1  = bih1[NXD + c0] + bhh1[NXD + c0];
  const float bIN1 = bih1[2 * NXD + c0];
  const float bHN1 = bhh1[2 * NXD + c0];
  const float bP1  = bp1[c0];
  const float bP2  = (wv < 2) ? bp2[c0] : 0.f;
  const float bWw  = (wv < 4) ? bw[c0] : 0.f;
  const float bQ0  = bq[c0];
  const float bQ1  = bq[NXD + c0];

  v8f h0s = {}, h1s = {};   // f32 hidden state, accumulator layout (cols c0)

  // ====================== init: x0 = tanh(Y0[0] Wq^T + bq) ======================
  for (int i = tid; i < 16 * NYD; i += 256) {
    const int m = i >> 5, n = i & 31;
    sE[m * LDE + n] = (__bf16)Y0[(size_t)(b0 + m) * NYD + n];
  }
  __syncthreads();
  {
    const v16bf aq  = lds_afrag(sE, LDE, 0, lane);
    const v16bf bL0 = gbl_bfrag(Wq, NYD, wv * 16, 0, lane);        // layer-0 cols
    const v16bf bL1 = gbl_bfrag(Wq, NYD, NXD + wv * 16, 0, lane);  // layer-1 cols
    v8f a0 = {}, a1 = {};
    a0 = wmma_bf(aq, bL0, a0);
    a1 = wmma_bf(aq, bL1, a1);
#pragma unroll
    for (int r = 0; r < 8; ++r) {
      h0s[r] = fast_tanh(a0[r] + bQ0);
      h1s[r] = fast_tanh(a1[r] + bQ1);
    }
  }
  __syncthreads();
#pragma unroll
  for (int r = 0; r < 8; ++r) {
    const int m = r + hi * 8;
    sX0[m * LDX + c0] = (__bf16)h0s[r];
    sX1[m * LDX + c0] = (__bf16)h1s[r];
    Xout[(size_t)(b0 + m) * NXD + c0] = h1s[r];   // X[0] = x0[-1] (layer 1)
  }
  __syncthreads();

  // readout: t = tanh(h1 Wp1^T + bp1); yhat = t Wp2^T + bp2
  auto phase_p1p2 = [&](float* yout_base) {
    v8f acc = {};
#pragma unroll
    for (int kc = 0; kc < 4; ++kc) {
      const v16bf a = lds_afrag(sX1, LDX, kc * 32, lane);
      acc = wmma_bf(a, fWp1[kc], acc);
    }
#pragma unroll
    for (int r = 0; r < 8; ++r) {
      const int m = r + hi * 8;
      sT[m * LDX + c0] = (__bf16)fast_tanh(acc[r] + bP1);
    }
    __syncthreads();
    if (wv < 2) {
      v8f ay = {};
#pragma unroll
      for (int kc = 0; kc < 4; ++kc) {
        const v16bf a = lds_afrag(sT, LDX, kc * 32, lane);
        ay = wmma_bf(a, fWp2[kc], ay);
      }
#pragma unroll
      for (int r = 0; r < 8; ++r) {
        const int m = r + hi * 8;
        const float yv = ay[r] + bP2;
        sYh[m * NYD + c0] = yv;
        yout_base[(size_t)(b0 + m) * NYD + c0] = yv;
      }
    }
    __syncthreads();
  };
  phase_p1p2(Yout);   // Yhat[0]

  // ====================== 512 sequential steps ======================
  for (int t = 0; t < TT; ++t) {
    const bool wup = (t < NWUP_);

    if (wup) {
      // ---- e = y_t - yhat (bf16 stage) ----
      const float* yrow = Y0 + (size_t)(1 + t) * NB * NYD;
      for (int i = tid; i < 16 * NYD; i += 256) {
        const int m = i >> 5, n = i & 31;
        sE[m * LDE + n] = (__bf16)(yrow[(size_t)(b0 + m) * NYD + n] - sYh[m * NYD + n]);
      }
      if (wv == 7 && t + 1 < NWUP_) {   // idle wave prefetches next y tile
        const float* np = Y0 + (size_t)(2 + t) * NB * NYD + (size_t)b0 * NYD;
        __builtin_prefetch(np + lane * 16, 0, 0);
      }
      __syncthreads();
      // ---- w = tanh([x0|e] Ww^T + bw) ----
      if (wv < 4) {
        v8f aw = {};
#pragma unroll
        for (int kc = 0; kc < 4; ++kc) {
          const v16bf a = lds_afrag(sX0, LDX, kc * 32, lane);
          aw = wmma_bf(a, fWw[kc], aw);
        }
        {
          const v16bf a = lds_afrag(sE, LDE, 0, lane);
          aw = wmma_bf(a, fWw[4], aw);
        }
        float* wrow = Wout + (size_t)t * NB * NWD;
#pragma unroll
        for (int r = 0; r < 8; ++r) {
          const int m = r + hi * 8;
          const float wval = fast_tanh(aw[r] + bWw);
          sW[m * LDW + c0] = (__bf16)wval;
          wrow[(size_t)(b0 + m) * NWD + c0] = wval;
        }
      }
      __syncthreads();
    }
    // ---- GRU layer 0: h0 = gru(w, h0). Horizon: w==0 -> skip gi WMMAs ----
    v8f accR = {}, accZ = {}, accIN = {}, accHN = {};
    if (wup) {
#pragma unroll
      for (int kc = 0; kc < 2; ++kc) {
        const v16bf a = lds_afrag(sW, LDW, kc * 32, lane);
        accR  = wmma_bf(a, fWih0[0][kc], accR);
        accZ  = wmma_bf(a, fWih0[1][kc], accZ);
        accIN = wmma_bf(a, fWih0[2][kc], accIN);
      }
    }
#pragma unroll
    for (int kc = 0; kc < 4; ++kc) {
      const v16bf a = lds_afrag(sX0, LDX, kc * 32, lane);
      accR  = wmma_bf(a, fWhh0[0][kc], accR);
      accZ  = wmma_bf(a, fWhh0[1][kc], accZ);
      accHN = wmma_bf(a, fWhh0[2][kc], accHN);
    }
    __syncthreads();   // all reads of sX0/sW done before overwrite
#pragma unroll
    for (int r = 0; r < 8; ++r) {
      const float rg = fast_sigmoid(accR[r] + bR0);
      const float zg = fast_sigmoid(accZ[r] + bZ0);
      const float ng = fast_tanh(accIN[r] + bIN0 + rg * (accHN[r] + bHN0));
      h0s[r] = (1.f - zg) * ng + zg * h0s[r];
      const int m = r + hi * 8;
      sX0[m * LDX + c0] = (__bf16)h0s[r];
    }
    __syncthreads();
    // ---- GRU layer 1: h1 = gru(h0_new, h1) ----
    v8f aR = {}, aZ = {}, aIN = {}, aHN = {};
#pragma unroll
    for (int kc = 0; kc < 4; ++kc) {
      const v16bf a0f = lds_afrag(sX0, LDX, kc * 32, lane);  // h0 new
      const v16bf a1f = lds_afrag(sX1, LDX, kc * 32, lane);  // h1 old
      aR  = wmma_bf(a0f, fWih1[0][kc], aR);
      aZ  = wmma_bf(a0f, fWih1[1][kc], aZ);
      aIN = wmma_bf(a0f, fWih1[2][kc], aIN);
      aR  = wmma_bf(a1f, fWhh1[0][kc], aR);
      aZ  = wmma_bf(a1f, fWhh1[1][kc], aZ);
      aHN = wmma_bf(a1f, fWhh1[2][kc], aHN);
    }
    __syncthreads();
    float* xrow = Xout + (size_t)(t + 1) * NB * NXD;
#pragma unroll
    for (int r = 0; r < 8; ++r) {
      const float rg = fast_sigmoid(aR[r] + bR1);
      const float zg = fast_sigmoid(aZ[r] + bZ1);
      const float ng = fast_tanh(aIN[r] + bIN1 + rg * (aHN[r] + bHN1));
      h1s[r] = (1.f - zg) * ng + zg * h1s[r];
      const int m = r + hi * 8;
      sX1[m * LDX + c0] = (__bf16)h1s[r];
      xrow[(size_t)(b0 + m) * NXD + c0] = h1s[r];   // X[t+1]
    }
    __syncthreads();
    // ---- readout ----
    phase_p1p2(Yout + (size_t)(t + 1) * NB * NYD);  // Yhat[t+1]
  }
}

extern "C" void kernel_launch(void* const* d_in, const int* in_sizes, int n_in,
                              void* d_out, int out_size, void* d_ws, size_t ws_size,
                              hipStream_t stream) {
  (void)in_sizes; (void)n_in; (void)out_size; (void)d_ws; (void)ws_size;
  const float* Y0   = (const float*)d_in[0];
  const float* Wq   = (const float*)d_in[1];
  const float* bq   = (const float*)d_in[2];
  const float* Wp1  = (const float*)d_in[3];
  const float* bp1  = (const float*)d_in[4];
  const float* Wp2  = (const float*)d_in[5];
  const float* bp2  = (const float*)d_in[6];
  const float* Ww   = (const float*)d_in[7];
  const float* bw   = (const float*)d_in[8];
  const float* Wih0 = (const float*)d_in[9];
  const float* Whh0 = (const float*)d_in[10];
  const float* bih0 = (const float*)d_in[11];
  const float* bhh0 = (const float*)d_in[12];
  const float* Wih1 = (const float*)d_in[13];
  const float* Whh1 = (const float*)d_in[14];
  const float* bih1 = (const float*)d_in[15];
  const float* bhh1 = (const float*)d_in[16];

  float* out  = (float*)d_out;
  float* Xout = out;                                   // (513,1024,128)
  float* Wout = Xout + (size_t)(TT + 1) * NB * NXD;    // (256,1024,64)
  float* Yout = Wout + (size_t)NWUP_ * NB * NWD;       // (513,1024,32)

  sinkbf_gru_rollout<<<NB / 16, 256, 0, stream>>>(
      Y0, Wq, bq, Wp1, bp1, Wp2, bp2, Ww, bw,
      Wih0, Whh0, bih0, bhh0, Wih1, Whh1, bih1, bhh1,
      Xout, Wout, Yout);
}
